// Joiner_23132693856322
// MI455X (gfx1250) — matrix operand order repaired and checked by reading, not compiled, not run on hardware
//
#include <hip/hip_runtime.h>

// ---------------------------------------------------------------------------
// RNN-T Joiner for gfx1250 (MI455X):
//   out[b,t,u,v] = sum_c tanh(enc[b,t,c] + pred[b,u,c]) * W[v,c] + bias[v]
// B=4 T=400 U=100 C=512 V=512.
// Strategy: f16 WMMA (v_wmma_f32_16x16x32_f16) with f32 accumulation.
//   - Pass 1: convert W to f16 in workspace (512 KB), stays L2-resident.
//   - Pass 2: per workgroup, build 64 activation rows (16 t x 4 u) in LDS as
//     f16 (fast tanh via v_exp_f32 + v_rcp_f32), then 8 waves run the 64x256
//     output tile as 16x16 WMMA tiles; bias added in epilogue; fp32 output
//     streamed with non-temporal stores (output = 327 MB, write-once).
// ---------------------------------------------------------------------------

typedef __attribute__((ext_vector_type(16))) _Float16 v16h;
typedef __attribute__((ext_vector_type(8)))  _Float16 v8h;
typedef __attribute__((ext_vector_type(4)))  _Float16 v4h;
typedef __attribute__((ext_vector_type(8)))  float    v8f;

#define B_ 4
#define T_ 400
#define U_ 100
#define C_ 512
#define V_ 512

// Padded LDS row stride in f16 elems: 520*2 = 1040 B -> rows land on distinct
// bank groups (1024 B stride would alias all 16 rows onto one bank).
#define AST 520

__device__ __forceinline__ float fast_tanh(float x) {
    // tanh(x) = 1 - 2/(exp(2x)+1); saturates to +/-1 correctly at the ends.
    float e = __expf(2.0f * x);                       // v_exp_f32 (trans)
    return 1.0f - 2.0f * __builtin_amdgcn_rcpf(e + 1.0f);  // v_rcp_f32
}

// ---- Pass 1: W (V x C, f32 row-major) -> f16, same layout ------------------
__global__ void wcvt_kernel(const float* __restrict__ W, _Float16* __restrict__ Wh) {
    int i = (blockIdx.x * blockDim.x + threadIdx.x) * 4;
    float4 w = *(const float4*)(W + i);
    v4h h = { (_Float16)w.x, (_Float16)w.y, (_Float16)w.z, (_Float16)w.w };
    *(v4h*)(Wh + i) = h;
}

// ---- Pass 2: joiner --------------------------------------------------------
__launch_bounds__(256)
__global__ void joiner_kernel(const float* __restrict__ enc,
                              const float* __restrict__ pred,
                              const _Float16* __restrict__ Wh,
                              const float* __restrict__ bias,
                              float* __restrict__ out) {
    __shared__ _Float16 Ash[64 * AST];   // 64 rows x 512 K (padded) = 65 KB

    // Block decode: vh in {0,1} selects 256-wide V half; (t16,u4) tile coords.
    int bid = blockIdx.x;
    const int vh  = bid & 1;  bid >>= 1;
    const int u4  = bid % 25; bid /= 25;
    const int t16 = bid % 25; bid /= 25;
    const int b   = bid;                  // 0..3
    const int t0  = t16 * 16;
    const int u0  = u4  * 4;

    // ---------------- Phase 1: activations -> LDS (f16) ----------------
    // 64 rows (row = tt*4 + uu) x 512 K, 4 elems per thread-iteration.
    {
        const float* encb = enc + ((size_t)b * T_ + t0) * C_;
        const float* prdb = pred + ((size_t)b * U_ + u0) * C_;
        #pragma unroll 4
        for (int i = 0; i < 32; ++i) {
            const int chunk = threadIdx.x + i * 256;   // 0..8191
            const int row   = chunk >> 7;              // 0..63
            const int c     = (chunk & 127) << 2;      // 0..508
            const int tt    = row >> 2;
            const int uu    = row & 3;
            float4 e = *(const float4*)(encb + (size_t)tt * C_ + c);
            float4 p = *(const float4*)(prdb + (size_t)uu * C_ + c);
            v4h h = { (_Float16)fast_tanh(e.x + p.x),
                      (_Float16)fast_tanh(e.y + p.y),
                      (_Float16)fast_tanh(e.z + p.z),
                      (_Float16)fast_tanh(e.w + p.w) };
            *(v4h*)(&Ash[row * AST + c]) = h;
        }
    }
    __syncthreads();

    // ---------------- Phase 2: WMMA GEMM ----------------
    const int wave = threadIdx.x >> 5;
    const int lane = threadIdx.x & 31;
    const int half = lane >> 4;          // 0: lanes 0-15, 1: lanes 16-31
    const int idx  = lane & 15;
    const int rt   = wave & 3;           // row tile 0..3 (16 rows each)
    const int chalf= wave >> 2;          // 0/1: 128-col half of this block's 256
    const int rowbase = rt * 16;
    const int colbase = vh * 256 + chalf * 128;

    v8f acc[8] = {};                     // 8 col-tiles of 16x16 f32

    #pragma unroll 2
    for (int ks = 0; ks < 16; ++ks) {    // K = 512 in steps of 32
        const int kb = ks * 32;

        // A fragment (16x32 f16): lane<16 holds row=idx K kb+0..7 / kb+16..23;
        // lane>=16 holds K kb+8..15 / kb+24..31  (ISA 16-bit A layout).
        const _Float16* ap = &Ash[(rowbase + idx) * AST + kb + half * 8];
        v8h a_lo = *(const v8h*)(ap);         // ds_load_b128
        v8h a_hi = *(const v8h*)(ap + 16);    // +32 bytes
        v16h afrag = __builtin_shufflevector(a_lo, a_hi,
            0,1,2,3,4,5,6,7,8,9,10,11,12,13,14,15);

        #pragma unroll
        for (int ct = 0; ct < 8; ++ct) {
            // B fragment (32x16 f16): lane = column n, K kb+half*8+0..7 /
            // kb+16+half*8+0..7 (same K order as A).  W row-major (V,C).
            const _Float16* bp = Wh + (size_t)(colbase + ct * 16 + idx) * C_
                                    + kb + half * 8;
            v8h b_lo = *(const v8h*)(bp);
            v8h b_hi = *(const v8h*)(bp + 16);
            v16h bfrag = __builtin_shufflevector(b_lo, b_hi,
                0,1,2,3,4,5,6,7,8,9,10,11,12,13,14,15);

            acc[ct] = __builtin_amdgcn_wmma_f32_16x16x32_f16(
                false, afrag, false, bfrag, (short)0, acc[ct], false, false);
        }
    }

    // ---------------- Epilogue: bias add + NT stores ----------------
    // C/D layout: VGPR j, lanes 0-15 -> M=j, lanes 16-31 -> M=j+8; N = idx.
    float* ob = out + (((size_t)b * T_ + t0) * U_ + u0) * V_;
    #pragma unroll
    for (int ct = 0; ct < 8; ++ct) {
        const int v  = colbase + ct * 16 + idx;
        const float bv = bias[v];
        #pragma unroll
        for (int j = 0; j < 8; ++j) {
            const int rl = rowbase + half * 8 + j;   // local row 0..63
            const int tt = rl >> 2;
            const int uu = rl & 3;
            __builtin_nontemporal_store(acc[ct][j] + bv,
                ob + (size_t)tt * (U_ * V_) + (size_t)uu * V_ + v);
        }
    }
}

extern "C" void kernel_launch(void* const* d_in, const int* in_sizes, int n_in,
                              void* d_out, int out_size, void* d_ws, size_t ws_size,
                              hipStream_t stream) {
    const float* enc  = (const float*)d_in[0];   // (B,T,C)
    const float* pred = (const float*)d_in[1];   // (B,U,C)
    const float* W    = (const float*)d_in[2];   // (V,C)
    const float* bias = (const float*)d_in[3];   // (V,)
    float* out = (float*)d_out;                  // (B,T,U,V) fp32
    _Float16* Wh = (_Float16*)d_ws;              // needs V_*C_*2 = 512 KB

    // Pass 1: W -> f16 (re-done every call; deterministic, graph-safe).
    wcvt_kernel<<<(V_ * C_ / 4) / 256, 256, 0, stream>>>(W, Wh);

    // Pass 2: 4 * 25 * 25 * 2 = 5000 workgroups of 256 threads (8 waves).
    joiner_kernel<<<B_ * (T_ / 16) * (U_ / 4) * 2, 256, 0, stream>>>(
        enc, pred, Wh, bias, out);
}
